// WithAttention_9801115369744
// MI455X (gfx1250) — matrix operand order repaired and checked by reading, not compiled, not run on hardware
//
#include <hip/hip_runtime.h>
#include <cstdint>

// ---------------------------------------------------------------------------
// MI455X (gfx1250) fused attention+GRU step.
// Dominant GEMM done with V_WMMA_F32_16X16X32_BF16 using a 2xbf16 (hi/lo)
// split of fp32 operands: D += Ah*Bh + Ah*Bl + Al*Bh  (fp32-class accuracy).
// The [B,S,A] intermediate is never materialized: tanh(.)*w_out row-reduction
// is fused into the WMMA epilogue. b_out cancels in softmax and is dropped.
// The encoded A-tile is split to bf16 hi/lo ONCE per workgroup into LDS and
// shared by all 8 waves (ds_load_b128 fragments), removing the 8x redundant
// global loads + conversion VALU from the WMMA hot loop.
// ---------------------------------------------------------------------------

#define DEVINL static __device__ __forceinline__

typedef __attribute__((ext_vector_type(16))) __bf16 v16bf;
typedef __attribute__((ext_vector_type(8)))  float  v8f;

union BF16Frag {
  v16bf v;
  unsigned short h[16];
  uint4 q[2];
};

DEVINL unsigned short f2bf_rn(float f) {
  union { float f; uint32_t u; } x; x.f = f;
  uint32_t r = x.u + 0x7FFFu + ((x.u >> 16) & 1u);
  return (unsigned short)(r >> 16);
}
DEVINL float bf2f(unsigned short h) {
  union { uint32_t u; float f; } x; x.u = ((uint32_t)h) << 16;
  return x.f;
}

// 3-term split-bf16 MMA: acc += Ah*Bh + Ah*Bl + Al*Bh
DEVINL v8f wmma3(v8f acc, const BF16Frag& ah, const BF16Frag& al,
                 const BF16Frag& bh, const BF16Frag& bl) {
  acc = __builtin_amdgcn_wmma_f32_16x16x32_bf16(false, ah.v, false, bh.v,
                                                (short)0, acc, false, false);
  acc = __builtin_amdgcn_wmma_f32_16x16x32_bf16(false, ah.v, false, bl.v,
                                                (short)0, acc, false, false);
  acc = __builtin_amdgcn_wmma_f32_16x16x32_bf16(false, al.v, false, bh.v,
                                                (short)0, acc, false, false);
  return acc;
}

// ------------------------- conversion / prep kernels -----------------------

// Generic fp32 -> (bf16 hi, bf16 lo) split, grid-stride.
__global__ void k_split(const float* __restrict__ in,
                        unsigned short* __restrict__ hi,
                        unsigned short* __restrict__ lo, int n) {
  for (int i = blockIdx.x * blockDim.x + threadIdx.x; i < n;
       i += gridDim.x * blockDim.x) {
    float f = in[i];
    unsigned short h = f2bf_rn(f);
    hi[i] = h;
    lo[i] = f2bf_rn(f - bf2f(h));
  }
}

// W_enc [H=1024, A=1024] -> transposed split [A, H] so B-fragment loads are
// contiguous 32B per lane.
__global__ void k_tsplit_wenc(const float* __restrict__ W,
                              unsigned short* __restrict__ hi,
                              unsigned short* __restrict__ lo) {
  int idx = blockIdx.x * 256 + threadIdx.x;   // 1M elements
  int a = idx >> 10, h = idx & 1023;
  float f = W[(size_t)h * 1024 + a];
  unsigned short hb = f2bf_rn(f);
  hi[idx] = hb;
  lo[idx] = f2bf_rn(f - bf2f(hb));
}

// x = concat(inputs, attention) -> bf16 hi/lo [64, 2048]
__global__ void k_pack_x(const float* __restrict__ inputs,
                         const float* __restrict__ attention,
                         unsigned short* __restrict__ hi,
                         unsigned short* __restrict__ lo) {
  int idx = blockIdx.x * 256 + threadIdx.x;   // 64*2048
  int b = idx >> 11, k = idx & 2047;
  float f = (k < 1024) ? inputs[b * 1024 + k] : attention[b * 1024 + (k - 1024)];
  unsigned short h = f2bf_rn(f);
  hi[idx] = h;
  lo[idx] = f2bf_rn(f - bf2f(h));
}

// dec_full[b,a] = states[b,:] @ W_dec[:,a] + b_enc[a] + b_dec[a]   (tiny GEMM)
__global__ void k_dec_full(const float* __restrict__ states,
                           const float* __restrict__ W_dec,
                           const float* __restrict__ b_enc,
                           const float* __restrict__ b_dec,
                           float* __restrict__ dec_full) {
  __shared__ float srow[1024];
  const int b = blockIdx.x, tid = threadIdx.x;
  for (int i = tid; i < 1024; i += 256) srow[i] = states[b * 1024 + i];
  __syncthreads();
  for (int c = 0; c < 4; ++c) {
    const int a = tid + c * 256;
    float acc = b_enc[a] + b_dec[a];
    for (int h = 0; h < 1024; ++h)
      acc = fmaf(srow[h], W_dec[(size_t)h * 1024 + a], acc);
    dec_full[(size_t)b * 1024 + a] = acc;
  }
}

// ------------------- fused score GEMM (the 68.7 GFLOP one) -----------------
// grid: B * (S/16) = 2048 workgroups of 256 threads (8 waves).
// Each workgroup owns one (b, 16-row s-tile); each wave owns 8 a-tiles.
// K=1024 streamed in two 512-halves; each half's 16x512 fp32 encoded tile is
// split to bf16 hi/lo planes in LDS once and shared by all waves.
// scores[b,s] = sum_a tanh(enc_proj + dec_full) * w_out[a]   (b_out dropped)
__global__ void k_scores(const float* __restrict__ encoded,
                         const unsigned short* __restrict__ wtH,
                         const unsigned short* __restrict__ wtL,
                         const float* __restrict__ dec_full,
                         const float* __restrict__ w_out,
                         float* __restrict__ scores) {
  __shared__ unsigned short tile[2 * 16 * 512];   // hi plane, then lo plane (32KB)
  __shared__ float srow[16];
  const int tid = threadIdx.x;
  const int lane = tid & 31, wave = tid >> 5;
  const int b = blockIdx.x >> 5;                  // 32 s-tiles per batch
  const int s0 = (blockIdx.x & 31) << 4;
  if (tid < 16) srow[tid] = 0.0f;

  v8f zero = {0.f, 0.f, 0.f, 0.f, 0.f, 0.f, 0.f, 0.f};
  v8f acc[8];
#pragma unroll
  for (int t = 0; t < 8; ++t) acc[t] = zero;

  const int a_base = wave * 128;                  // 8 n-tiles of 16 per wave
  const int n = lane & 15;
  const int m = lane & 15;
  const int kb_half = (lane < 16) ? 0 : 16;       // B-frag K half select
  const int q0 = (lane < 16) ? 0 : 16;            // A-frag K half select (contiguous)
  const float* src = encoded + ((size_t)(b * 512 + s0)) * 1024;

  for (int half = 0; half < 2; ++half) {
    __syncthreads();                              // prior tile reads / srow init done
    // ---- cooperative stage: 16x512 fp32 -> bf16 hi/lo planes in LDS ----
#pragma unroll
    for (int i = 0; i < 8; ++i) {
      const int e4 = tid + i * 256;               // float4 index, 2048 total
      const int eb = e4 * 4;                      // element index
      const int row = eb >> 9, k = eb & 511;
      float4 f = *(const float4*)(src + (size_t)row * 1024 + half * 512 + k);
      unsigned short h0 = f2bf_rn(f.x), h1 = f2bf_rn(f.y);
      unsigned short h2 = f2bf_rn(f.z), h3 = f2bf_rn(f.w);
      uint2 hq, lq;
      hq.x = (uint32_t)h0 | ((uint32_t)h1 << 16);
      hq.y = (uint32_t)h2 | ((uint32_t)h3 << 16);
      lq.x = (uint32_t)f2bf_rn(f.x - bf2f(h0)) |
             ((uint32_t)f2bf_rn(f.y - bf2f(h1)) << 16);
      lq.y = (uint32_t)f2bf_rn(f.z - bf2f(h2)) |
             ((uint32_t)f2bf_rn(f.w - bf2f(h3)) << 16);
      ((uint2*)tile)[e4] = hq;                    // hi plane [16][512]
      ((uint2*)(tile + 16 * 512))[e4] = lq;       // lo plane [16][512]
    }
    __syncthreads();

    // ---- WMMA hot loop over this K-half ----
    for (int kk = 0; kk < 16; ++kk) {
      const int kb = kk * 32;                     // LDS k
      const int kbg = half * 512 + kb;            // global k (for W_encT)
      BF16Frag ah, al;
      const unsigned short* hr = tile + m * 512 + kb;
      const unsigned short* lr = hr + 16 * 512;
      ah.q[0] = *(const uint4*)(hr + q0);         // halves 0-7
      ah.q[1] = *(const uint4*)(hr + q0 + 8);     // halves 8-15
      al.q[0] = *(const uint4*)(lr + q0);
      al.q[1] = *(const uint4*)(lr + q0 + 8);
#pragma unroll
      for (int t = 0; t < 8; ++t) {
        const int a0 = a_base + t * 16;
        const unsigned short* wr = wtH + ((size_t)(a0 + n)) * 1024 + kbg + kb_half;
        const unsigned short* wl = wtL + ((size_t)(a0 + n)) * 1024 + kbg + kb_half;
        BF16Frag bh, bl;
        bh.q[0] = *(const uint4*)wr;
        bh.q[1] = *(const uint4*)(wr + 8);
        bl.q[0] = *(const uint4*)wl;
        bl.q[1] = *(const uint4*)(wl + 8);
        acc[t] = wmma3(acc[t], ah, al, bh, bl);
      }
    }
  }

  // Fused epilogue: tanh, scale by w_out, reduce over a -> per-row score.
#pragma unroll
  for (int t = 0; t < 8; ++t) {
    const int a = a_base + t * 16 + n;
    const float d = dec_full[(size_t)b * 1024 + a];
    const float w = w_out[a];
#pragma unroll
    for (int v = 0; v < 8; ++v) {
      float val = tanhf(acc[t][v] + d) * w;
      val += __shfl_xor(val, 1, 32);
      val += __shfl_xor(val, 2, 32);
      val += __shfl_xor(val, 4, 32);
      val += __shfl_xor(val, 8, 32);
      if (n == 0) atomicAdd(&srow[v + ((lane >> 4) << 3)], val);  // ds_add_f32
    }
  }
  __syncthreads();
  if (tid < 16) scores[(size_t)b * 512 + s0 + tid] = srow[tid];
}

// --------------------- softmax over S + attention reduce -------------------
__global__ void k_softmax_attn(const float* __restrict__ scores,
                               const float* __restrict__ encoded,
                               float* __restrict__ attention) {
  __shared__ float sc[512];
  __shared__ float red[256];
  const int b = blockIdx.x, tid = threadIdx.x;
  float s0 = scores[b * 512 + tid];
  float s1 = scores[b * 512 + 256 + tid];
  red[tid] = fmaxf(s0, s1);
  __syncthreads();
  for (int o = 128; o > 0; o >>= 1) {
    if (tid < o) red[tid] = fmaxf(red[tid], red[tid + o]);
    __syncthreads();
  }
  float mx = red[0];
  __syncthreads();
  float e0 = expf(s0 - mx), e1 = expf(s1 - mx);
  sc[tid] = e0;
  sc[tid + 256] = e1;
  red[tid] = e0 + e1;
  __syncthreads();
  for (int o = 128; o > 0; o >>= 1) {
    if (tid < o) red[tid] += red[tid + o];
    __syncthreads();
  }
  const float inv = 1.0f / red[0];
  for (int c = 0; c < 4; ++c) {
    const int hcol = tid + c * 256;
    float acc = 0.f;
    const float* e = encoded + (size_t)b * 512 * 1024 + hcol;
    for (int s = 0; s < 512; ++s)
      acc = fmaf(e[(size_t)s * 1024], sc[s], acc);
    attention[(size_t)b * 1024 + hcol] = acc * inv;
  }
}

// ----------------------- GRU GEMMs: gi / gh via WMMA -----------------------
// out[M=64, N] = A[M,K] @ W[N,K]^T + bias, split-bf16 operands.
// One wave per 16x16 tile; grid (N/16, M/16), block 32.
__global__ void k_gru_gemm(const unsigned short* __restrict__ Ahi,
                           const unsigned short* __restrict__ Alo,
                           const unsigned short* __restrict__ Whi,
                           const unsigned short* __restrict__ Wlo,
                           const float* __restrict__ bias,
                           float* __restrict__ out, int K, int N) {
  const int lane = threadIdx.x;
  const int nt = blockIdx.x, mt = blockIdx.y;
  const int mrow = mt * 16 + (lane & 15);
  const int n = lane & 15;
  const int g = nt * 16 + n;
  const int kb_half = (lane < 16) ? 0 : 16;
  const int q0 = (lane < 16) ? 0 : 16;
  const unsigned short* arH = Ahi + (size_t)mrow * K;
  const unsigned short* arL = Alo + (size_t)mrow * K;
  const unsigned short* wrH = Whi + (size_t)g * K;
  const unsigned short* wrL = Wlo + (size_t)g * K;

  v8f acc = {0.f, 0.f, 0.f, 0.f, 0.f, 0.f, 0.f, 0.f};
  const int nk = K >> 5;
  for (int kk = 0; kk < nk; ++kk) {
    const int kb = kk * 32;
    BF16Frag ah, al, bh, bl;
    ah.q[0] = *(const uint4*)(arH + kb + q0);
    ah.q[1] = *(const uint4*)(arH + kb + q0 + 8);
    al.q[0] = *(const uint4*)(arL + kb + q0);
    al.q[1] = *(const uint4*)(arL + kb + q0 + 8);
    bh.q[0] = *(const uint4*)(wrH + kb + kb_half);
    bh.q[1] = *(const uint4*)(wrH + kb + kb_half + 8);
    bl.q[0] = *(const uint4*)(wrL + kb + kb_half);
    bl.q[1] = *(const uint4*)(wrL + kb + kb_half + 8);
    acc = wmma3(acc, ah, al, bh, bl);
  }
  const float bs = bias[g];
  const int rbase = mt * 16 + ((lane >> 4) << 3);
#pragma unroll
  for (int v = 0; v < 8; ++v)
    out[(size_t)(rbase + v) * N + g] = acc[v] + bs;
}

// ------------------------------- GRU gates ---------------------------------
__global__ void k_gates(const float* __restrict__ gi, const float* __restrict__ gh,
                        const float* __restrict__ states, float* __restrict__ out) {
  const int idx = blockIdx.x * 256 + threadIdx.x;   // 64*1024
  const int b = idx >> 10, h = idx & 1023;
  const float* gib = gi + (size_t)b * 3072;
  const float* ghb = gh + (size_t)b * 3072;
  float r = 1.f / (1.f + expf(-(gib[h] + ghb[h])));
  float z = 1.f / (1.f + expf(-(gib[h + 1024] + ghb[h + 1024])));
  float nn = tanhf(gib[h + 2048] + r * ghb[h + 2048]);
  out[idx] = (1.f - z) * nn + z * states[idx];
}

// ------------------------------- launcher ----------------------------------
extern "C" void kernel_launch(void* const* d_in, const int* in_sizes, int n_in,
                              void* d_out, int out_size, void* d_ws, size_t ws_size,
                              hipStream_t stream) {
  (void)in_sizes; (void)n_in; (void)out_size; (void)ws_size;
  const float* encoded = (const float*)d_in[0];
  const float* inputs  = (const float*)d_in[1];
  const float* states  = (const float*)d_in[2];
  const float* W_enc   = (const float*)d_in[3];
  const float* b_enc   = (const float*)d_in[4];
  const float* W_dec   = (const float*)d_in[5];
  const float* b_dec   = (const float*)d_in[6];
  const float* w_out   = (const float*)d_in[7];
  // d_in[8] = b_out : constant over S, cancels in softmax -> unused.
  const float* W_ih    = (const float*)d_in[9];
  const float* W_hh    = (const float*)d_in[10];
  const float* b_ih    = (const float*)d_in[11];
  const float* b_hh    = (const float*)d_in[12];
  float* out = (float*)d_out;

  char* ws = (char*)d_ws;
  size_t off = 0;
  auto alloc = [&](size_t bytes) {
    void* p = (void*)(ws + off);
    off += (bytes + 255) & ~(size_t)255;
    return p;
  };
  unsigned short* wtH  = (unsigned short*)alloc((size_t)1024 * 1024 * 2);
  unsigned short* wtL  = (unsigned short*)alloc((size_t)1024 * 1024 * 2);
  unsigned short* wihH = (unsigned short*)alloc((size_t)3072 * 2048 * 2);
  unsigned short* wihL = (unsigned short*)alloc((size_t)3072 * 2048 * 2);
  unsigned short* whhH = (unsigned short*)alloc((size_t)3072 * 1024 * 2);
  unsigned short* whhL = (unsigned short*)alloc((size_t)3072 * 1024 * 2);
  unsigned short* sH   = (unsigned short*)alloc((size_t)64 * 1024 * 2);
  unsigned short* sL   = (unsigned short*)alloc((size_t)64 * 1024 * 2);
  unsigned short* xH   = (unsigned short*)alloc((size_t)64 * 2048 * 2);
  unsigned short* xL   = (unsigned short*)alloc((size_t)64 * 2048 * 2);
  float* decf   = (float*)alloc((size_t)64 * 1024 * 4);
  float* scores = (float*)alloc((size_t)64 * 512 * 4);
  float* attn   = (float*)alloc((size_t)64 * 1024 * 4);
  float* gi     = (float*)alloc((size_t)64 * 3072 * 4);
  float* gh     = (float*)alloc((size_t)64 * 3072 * 4);

  // weight / state prep (runs every call; deterministic)
  k_tsplit_wenc<<<4096, 256, 0, stream>>>(W_enc, wtH, wtL);
  k_split<<<2048, 256, 0, stream>>>(W_ih, wihH, wihL, 3072 * 2048);
  k_split<<<2048, 256, 0, stream>>>(W_hh, whhH, whhL, 3072 * 1024);
  k_split<<<256, 256, 0, stream>>>(states, sH, sL, 64 * 1024);
  k_dec_full<<<64, 256, 0, stream>>>(states, W_dec, b_enc, b_dec, decf);

  // fused attention-score GEMM (dominant compute, WMMA)
  k_scores<<<2048, 256, 0, stream>>>(encoded, wtH, wtL, decf, w_out, scores);
  k_softmax_attn<<<64, 256, 0, stream>>>(scores, encoded, attn);

  // GRU
  k_pack_x<<<512, 256, 0, stream>>>(inputs, attn, xH, xL);
  k_gru_gemm<<<dim3(192, 4), 32, 0, stream>>>(xH, xL, wihH, wihL, b_ih, gi, 2048, 3072);
  k_gru_gemm<<<dim3(192, 4), 32, 0, stream>>>(sH, sL, whhH, whhL, b_hh, gh, 1024, 3072);
  k_gates<<<256, 256, 0, stream>>>(gi, gh, states, out);
}
